// BiLSTM_CRF_51221779972591
// MI455X (gfx1250) — matrix operand order repaired and checked by reading, not compiled, not run on hardware
//
#include <hip/hip_runtime.h>

typedef _Float16 f16;
typedef __attribute__((ext_vector_type(16))) _Float16     v16h;
typedef __attribute__((ext_vector_type(8)))  float        v8f;
typedef __attribute__((ext_vector_type(8)))  unsigned int v8u;

#define SEQ    256
#define BATCH  64
#define HID    512
#define GN     2048   // 4*HID (gates i,f,g,o)
#define EMBP   320    // 300 padded to multiple of 32
#define NTAGS  20
#define TSTART 18
#define TSTOP  19

union Frag { v8u u; v16h h; };

__device__ __forceinline__ v8f wmma16x16x32(const Frag& a, const Frag& b, v8f c) {
  // D = A(16x32 f16) * B(32x16 f16) + C(16x16 f32)
  return __builtin_amdgcn_wmma_f32_16x16x32_f16(false, a.h, false, b.h,
                                                (short)0, c, false, false);
}

__device__ __forceinline__ float sigm(float x) { return 1.f / (1.f + __expf(-x)); }

// --- CDNA5 async LDS DMA (ASYNCcnt-tracked). Each lane copies 16B. -----------
__device__ __forceinline__ void async_cp16(unsigned lds_byte_addr, const f16* gsrc) {
  asm volatile("global_load_async_to_lds_b128 %0, %1, off"
               :: "v"(lds_byte_addr), "v"((unsigned long long)(size_t)gsrc)
               : "memory");
}
__device__ __forceinline__ void wait_async0() {
  asm volatile("s_wait_asynccnt 0" ::: "memory");
}
// Low 32 bits of a generic pointer to __shared__ == LDS byte address (ISA §10.2)
__device__ __forceinline__ unsigned lds_addr32(const void* p) {
  return (unsigned)(size_t)p;
}

// ---------------------------------------------------------------------------
// Embedding gather: x0[p][0..299] = emb[sent[p]][:], pad to 320 with zeros. f16 out.
// ---------------------------------------------------------------------------
__global__ void embed_gather(const int* __restrict__ sent,
                             const float* __restrict__ emb,
                             f16* __restrict__ x0) {
  const int p = blockIdx.x;               // 0..16383  (p = s*BATCH + b)
  const int idx = sent[p];
  const float* src = emb + (size_t)idx * 300;
  for (int e = threadIdx.x; e < EMBP; e += 256)
    x0[(size_t)p * EMBP + e] = (e < 300) ? (f16)src[e] : (f16)0.f;
}

// ---------------------------------------------------------------------------
// Weight convert f32 -> f16 with K padding (K..Kp zero) and N padding (N..Npad zero)
// ---------------------------------------------------------------------------
__global__ void conv_w(const float* __restrict__ in, f16* __restrict__ out,
                       int N, int K, int Kp, int Npad) {
  const int i = blockIdx.x * 256 + threadIdx.x;
  const int total = Npad * Kp;
  if (i >= total) return;
  const int n = i / Kp, k = i % Kp;
  out[i] = (n < N && k < K) ? (f16)in[(size_t)n * K + k] : (f16)0.f;
}

// ---------------------------------------------------------------------------
// Projection GEMM with async-LDS double buffering.
// C[M,N](f16) = A[M,K(f16,lda)] * B[N,K(f16,ldb)]^T + bias (bias length >= N).
// Block = 256 threads (8 waves), 128x128 tile; wave = 32x64 (2 A-frags x 4 B-frags,
// 8 WMMA per K-slice). A/B K-slices (8KB each) staged via global_load_async_to_lds_b128
// into ping-pong buffers; stage k+1 DMA overlaps stage k WMMAs (s_wait_asynccnt).
// M % 128 == 0, N % 128 == 0, K % 32 == 0.
// ---------------------------------------------------------------------------
__global__ __launch_bounds__(256)
void gemm_nt_async_wmma(const f16* __restrict__ A, const f16* __restrict__ Bw,
                        const float* __restrict__ bias, f16* __restrict__ Ch,
                        int N, int K, int lda, int ldb) {
  __shared__ __align__(16) f16 As[2][128 * 32];
  __shared__ __align__(16) f16 Bs[2][128 * 32];
  const int tid  = threadIdx.x;
  const int w    = tid >> 5, lane = tid & 31;
  const int msub = w >> 1,   nsub = w & 1;      // 4 x 2 wave grid
  const int half = lane >> 4, lr  = lane & 15;
  const int m0 = blockIdx.x * 128, n0 = blockIdx.y * 128;

  const unsigned ldsA[2] = { lds_addr32(&As[0][0]), lds_addr32(&As[1][0]) };
  const unsigned ldsB[2] = { lds_addr32(&Bs[0][0]), lds_addr32(&Bs[1][0]) };

  v8f acc[2][4] = {};

  // Stage one 128x32 K-slice of A and B: 512 chunks of 16B each, 2 per thread.
  auto issue = [&](int buf, int k0) {
#pragma unroll
    for (int j = 0; j < 2; ++j) {
      const int c   = tid + j * 256;        // 0..511
      const int row = c >> 2;               // 4 x 16B chunks per 32-f16 row
      const int co  = (c & 3) * 8;          // f16 elements
      async_cp16(ldsA[buf] + (unsigned)(c << 4),
                 A + (size_t)(m0 + row) * lda + k0 + co);
      async_cp16(ldsB[buf] + (unsigned)(c << 4),
                 Bw + (size_t)(n0 + row) * ldb + k0 + co);
    }
  };

  const int nk = K / 32;
  issue(0, 0);
  for (int kc = 0; kc < nk; ++kc) {
    wait_async0();        // my stage-kc DMA complete (counter per wave)
    __syncthreads();      // everyone's stage-kc DMA complete
    if (kc + 1 < nk) issue((kc + 1) & 1, (kc + 1) * 32);

    const f16* as = &As[kc & 1][0];
    const f16* bs = &Bs[kc & 1][0];
    Frag a0, a1;
#pragma unroll
    for (int v = 0; v < 8; ++v) {
      const int k = half * 16 + v * 2;                         // A 16x32 layout
      a0.u[v] = *(const unsigned int*)&as[(msub * 32      + lr) * 32 + k];
      a1.u[v] = *(const unsigned int*)&as[(msub * 32 + 16 + lr) * 32 + k];
    }
#pragma unroll
    for (int bi = 0; bi < 4; ++bi) {
      Frag b;
#pragma unroll
      for (int v = 0; v < 8; ++v) {
        const int k = half * 16 + v * 2;                       // B 32x16 layout
        b.u[v] = *(const unsigned int*)&bs[(nsub * 64 + bi * 16 + lr) * 32 + k];
      }
      acc[0][bi] = wmma16x16x32(a0, b, acc[0][bi]);
      acc[1][bi] = wmma16x16x32(a1, b, acc[1][bi]);
    }
  }
#pragma unroll
  for (int ar = 0; ar < 2; ++ar)
#pragma unroll
    for (int bi = 0; bi < 4; ++bi)
#pragma unroll
      for (int r = 0; r < 8; ++r) {
        const int m = m0 + msub * 32 + ar * 16 + r + 8 * half;
        const int n = n0 + nsub * 64 + bi * 16 + lr;
        Ch[(size_t)m * N + n] = (f16)(acc[ar][bi][r] + bias[n]);
      }
}

// ---------------------------------------------------------------------------
// Small NT GEMM (used for the FC epilogue, N=64): 64x64 block tile, 8 waves.
// ---------------------------------------------------------------------------
__global__ __launch_bounds__(256)
void gemm_nt_wmma(const f16* __restrict__ A, const f16* __restrict__ Bw,
                  const float* __restrict__ bias, int biasN,
                  float* __restrict__ Cf, int N, int K, int lda, int ldb) {
  __shared__ __align__(16) f16 As[64 * 32];
  __shared__ __align__(16) f16 Bs[64 * 32];
  const int tid  = threadIdx.x;
  const int w    = tid >> 5, lane = tid & 31;
  const int msub = w >> 1,   nsub = w & 1;
  const int half = lane >> 4, lr  = lane & 15;
  const int m0 = blockIdx.x * 64, n0 = blockIdx.y * 64;

  v8f acc0 = {}, acc1 = {};
  const int ldr = tid >> 2;        // 0..63
  const int ldc = (tid & 3) * 8;   // 0,8,16,24  (16B chunks)

  for (int k0 = 0; k0 < K; k0 += 32) {
    *(uint4*)&As[ldr * 32 + ldc] =
        *(const uint4*)&A[(size_t)(m0 + ldr) * lda + k0 + ldc];
    *(uint4*)&Bs[ldr * 32 + ldc] =
        *(const uint4*)&Bw[(size_t)(n0 + ldr) * ldb + k0 + ldc];
    if (k0 + 32 < K)
      __builtin_prefetch(&A[(size_t)(m0 + ldr) * lda + k0 + 32 + ldc], 0, 0);
    __syncthreads();

    Frag a;
#pragma unroll
    for (int v = 0; v < 8; ++v) {
      const int k = half * 16 + v * 2;
      a.u[v] = *(const unsigned int*)&As[(msub * 16 + lr) * 32 + k];
    }
    Frag b0, b1;
#pragma unroll
    for (int v = 0; v < 8; ++v) {
      const int k = half * 16 + v * 2;
      b0.u[v] = *(const unsigned int*)&Bs[(nsub * 32      + lr) * 32 + k];
      b1.u[v] = *(const unsigned int*)&Bs[(nsub * 32 + 16 + lr) * 32 + k];
    }
    acc0 = wmma16x16x32(a, b0, acc0);
    acc1 = wmma16x16x32(a, b1, acc1);
    __syncthreads();
  }
#pragma unroll
  for (int r = 0; r < 8; ++r) {
    const int m  = m0 + msub * 16 + r + 8 * half;
    const int n  = n0 + nsub * 32 + lr;
    const int n1 = n + 16;
    Cf[(size_t)m * N + n]  = acc0[r] + ((n  < biasN) ? bias[n]  : 0.f);
    Cf[(size_t)m * N + n1] = acc1[r] + ((n1 < biasN) ? bias[n1] : 0.f);
  }
}

// ---------------------------------------------------------------------------
// Persistent bidirectional LSTM recurrence for one layer.
// Grid = 8 blocks: blockIdx>>2 = direction, blockIdx&3 = batch slab of 16 rows.
// c-state lives in accumulator VGPRs across all 256 steps; h lives in LDS.
// ---------------------------------------------------------------------------
__global__ __launch_bounds__(256)
void lstm_recur_wmma(const f16* __restrict__ pre_f, const f16* __restrict__ pre_b,
                     const f16* __restrict__ whh_f, const f16* __restrict__ whh_b,
                     f16* __restrict__ hseq) {
  __shared__ __align__(16) f16 hbuf[16 * HID];  // 16 KB: this slab's h_prev
  const int dir  = blockIdx.x >> 2;
  const int bblk = blockIdx.x & 3;
  const f16* pre = dir ? pre_b : pre_f;
  const f16* whh = dir ? whh_b : whh_f;
  const int dirOff = dir * HID;

  const int tid  = threadIdx.x;
  const int wn   = tid >> 5, lane = tid & 31;
  const int half = lane >> 4, lr  = lane & 15;

  for (int i = tid; i < 16 * HID; i += 256) hbuf[i] = (f16)0.f;
  __syncthreads();

  v8f cst[4] = {};    // persistent cell state: 4 col-subtiles x 8 f32
  v8f acc[4][4];      // [gate i/f/g/o][col-subtile]

  for (int step = 0; step < SEQ; ++step) {
    const int s = dir ? (SEQ - 1 - step) : step;
    const f16* preS = pre + ((size_t)s * BATCH + bblk * 16) * GN;

    // init accumulators with input-projection pre-activations (bias folded in)
#pragma unroll
    for (int g = 0; g < 4; ++g) {
#pragma unroll
      for (int t = 0; t < 4; ++t) {
        const int ncol = g * HID + wn * 64 + t * 16 + lr;
        v8f a;
#pragma unroll
        for (int r = 0; r < 8; ++r)
          a[r] = (float)preS[(size_t)(r + 8 * half) * GN + ncol];
        acc[g][t] = a;
      }
    }

    // + h_prev @ Whh^T  (K = 512, 16 chunks of 32, 16 WMMA per chunk per wave)
#pragma unroll 1
    for (int kc = 0; kc < 16; ++kc) {
      Frag a;
#pragma unroll
      for (int v = 0; v < 8; ++v) {
        const int k = kc * 32 + half * 16 + v * 2;
        a.u[v] = *(const unsigned int*)&hbuf[lr * HID + k];
      }
#pragma unroll
      for (int g = 0; g < 4; ++g) {
#pragma unroll
        for (int t = 0; t < 4; ++t) {
          Frag b;
          const int n = g * HID + wn * 64 + t * 16 + lr;
#pragma unroll
          for (int v = 0; v < 8; ++v) {
            const int k = kc * 32 + half * 16 + v * 2;
            b.u[v] = *(const unsigned int*)&whh[(size_t)n * HID + k];
          }
          acc[g][t] = wmma16x16x32(a, b, acc[g][t]);
        }
      }
    }
    __syncthreads();  // all hbuf reads done before overwrite

    // elementwise LSTM cell + write h (LDS for next step, global h-seq)
    f16* outS = hseq + ((size_t)s * BATCH + bblk * 16) * (2 * HID) + dirOff;
#pragma unroll
    for (int t = 0; t < 4; ++t) {
#pragma unroll
      for (int r = 0; r < 8; ++r) {
        const float iv = acc[0][t][r], fv = acc[1][t][r];
        const float gv = acc[2][t][r], ov = acc[3][t][r];
        const float c  = sigm(fv) * cst[t][r] + sigm(iv) * tanhf(gv);
        cst[t][r] = c;
        const float h = sigm(ov) * tanhf(c);
        const int row = r + 8 * half;              // local batch row 0..15
        const int col = wn * 64 + t * 16 + lr;     // h column 0..511
        const f16 hh = (f16)h;
        hbuf[row * HID + col] = hh;
        outS[(size_t)row * (2 * HID) + col] = hh;
      }
    }
    __syncthreads();
  }
}

// ---------------------------------------------------------------------------
// Viterbi decode: one block of 64 threads, thread b handles batch row b.
// ---------------------------------------------------------------------------
__global__ __launch_bounds__(64)
void viterbi_decode(const float* __restrict__ feats,  // [SEQ*BATCH, 64] cols 0..19
                    const float* __restrict__ trans,  // [20,20] row=next, col=prev
                    unsigned char* __restrict__ bp,   // [SEQ,BATCH,20]
                    float* __restrict__ out) {        // [BATCH,SEQ]
  __shared__ float tr[NTAGS * NTAGS];
  const int tid = threadIdx.x;
  for (int i = tid; i < NTAGS * NTAGS; i += 64) tr[i] = trans[i];
  __syncthreads();

  const int b = tid;
  float v[NTAGS];
#pragma unroll
  for (int t = 0; t < NTAGS; ++t) v[t] = (t == TSTART) ? 0.f : -10000.f;

  for (int s = 0; s < SEQ; ++s) {
    const float* f = feats + ((size_t)s * BATCH + b) * 64;
    float nv[NTAGS];
    for (int nx = 0; nx < NTAGS; ++nx) {
      float best = -3.4e38f; int bi = 0;
      for (int pv = 0; pv < NTAGS; ++pv) {
        const float sc = v[pv] + tr[nx * NTAGS + pv];
        if (sc > best) { best = sc; bi = pv; }   // first-max, matches argmax
      }
      nv[nx] = best + f[nx];
      bp[((size_t)s * BATCH + b) * NTAGS + nx] = (unsigned char)bi;
    }
#pragma unroll
    for (int t = 0; t < NTAGS; ++t) v[t] = nv[t];
  }

  float best = -3.4e38f; int tag = 0;
  for (int t = 0; t < NTAGS; ++t) {
    const float sc = v[t] + tr[TSTOP * NTAGS + t];
    if (sc > best) { best = sc; tag = t; }
  }
  for (int s = SEQ - 1; s >= 0; --s) {
    out[(size_t)b * SEQ + s] = (float)tag;
    tag = bp[((size_t)s * BATCH + b) * NTAGS + tag];
  }
}

// ---------------------------------------------------------------------------
extern "C" void kernel_launch(void* const* d_in, const int* in_sizes, int n_in,
                              void* d_out, int out_size, void* d_ws, size_t ws_size,
                              hipStream_t stream) {
  (void)in_sizes; (void)n_in; (void)out_size; (void)ws_size;
  const int*   sent = (const int*)  d_in[0];
  const float* emb  = (const float*)d_in[1];
  const float* W_ih[2][2] = {{(const float*)d_in[2],  (const float*)d_in[5]},
                             {(const float*)d_in[8],  (const float*)d_in[11]}};
  const float* W_hh[2][2] = {{(const float*)d_in[3],  (const float*)d_in[6]},
                             {(const float*)d_in[9],  (const float*)d_in[12]}};
  const float* Bias[2][2] = {{(const float*)d_in[4],  (const float*)d_in[7]},
                             {(const float*)d_in[10], (const float*)d_in[13]}};
  const float* fcW   = (const float*)d_in[14];
  const float* fcB   = (const float*)d_in[15];
  const float* trans = (const float*)d_in[16];
  float* out = (float*)d_out;

  // workspace carve-out (256B aligned)
  char* ws = (char*)d_ws;
  size_t off = 0;
  auto take = [&](size_t bytes) -> char* {
    char* p = ws + off; off += (bytes + 255) & ~(size_t)255; return p;
  };
  f16* x0 = (f16*)take((size_t)SEQ * BATCH * EMBP * 2);
  f16* wih0[2] = {(f16*)take((size_t)GN * EMBP * 2), (f16*)take((size_t)GN * EMBP * 2)};
  f16* whhw[2][2];
  for (int l = 0; l < 2; ++l)
    for (int d = 0; d < 2; ++d) whhw[l][d] = (f16*)take((size_t)GN * HID * 2);
  f16* wih1[2] = {(f16*)take((size_t)GN * 1024 * 2), (f16*)take((size_t)GN * 1024 * 2)};
  f16* wfc   = (f16*)take((size_t)64 * 1024 * 2);
  f16* preF  = (f16*)take((size_t)SEQ * BATCH * GN * 2);
  f16* preB  = (f16*)take((size_t)SEQ * BATCH * GN * 2);
  f16* h0    = (f16*)take((size_t)SEQ * BATCH * 1024 * 2);
  f16* h1    = (f16*)take((size_t)SEQ * BATCH * 1024 * 2);
  float* feats = (float*)take((size_t)SEQ * BATCH * 64 * 4);
  unsigned char* bp = (unsigned char*)take((size_t)SEQ * BATCH * NTAGS);

  // 1) embedding gather (f32 -> f16, pad 300->320)
  embed_gather<<<SEQ * BATCH, 256, 0, stream>>>(sent, emb, x0);

  // 2) weight conversion
  auto conv = [&](const float* src, f16* dst, int N, int K, int Kp, int Npad) {
    const int total = Npad * Kp;
    conv_w<<<(total + 255) / 256, 256, 0, stream>>>(src, dst, N, K, Kp, Npad);
  };
  conv(W_ih[0][0], wih0[0], GN, 300, EMBP, GN);
  conv(W_ih[0][1], wih0[1], GN, 300, EMBP, GN);
  for (int l = 0; l < 2; ++l)
    for (int d = 0; d < 2; ++d) conv(W_hh[l][d], whhw[l][d], GN, HID, HID, GN);
  conv(W_ih[1][0], wih1[0], GN, 1024, 1024, GN);
  conv(W_ih[1][1], wih1[1], GN, 1024, 1024, GN);
  conv(fcW, wfc, NTAGS, 1024, 1024, 64);

  const dim3 blk(256);
  const dim3 gP(SEQ * BATCH / 128, GN / 128);   // 128 x 16 blocks
  // 3) layer-0 input projections (M=16384, N=2048, K=320), async-LDS double buffered
  gemm_nt_async_wmma<<<gP, blk, 0, stream>>>(x0, wih0[0], Bias[0][0], preF, GN, EMBP, EMBP, EMBP);
  gemm_nt_async_wmma<<<gP, blk, 0, stream>>>(x0, wih0[1], Bias[0][1], preB, GN, EMBP, EMBP, EMBP);
  // 4) layer-0 recurrence (both directions, 4 batch slabs each)
  lstm_recur_wmma<<<8, 256, 0, stream>>>(preF, preB, whhw[0][0], whhw[0][1], h0);
  // 5) layer-1 projections (K=1024) reuse pre buffers
  gemm_nt_async_wmma<<<gP, blk, 0, stream>>>(h0, wih1[0], Bias[1][0], preF, GN, 1024, 1024, 1024);
  gemm_nt_async_wmma<<<gP, blk, 0, stream>>>(h0, wih1[1], Bias[1][1], preB, GN, 1024, 1024, 1024);
  // 6) layer-1 recurrence
  lstm_recur_wmma<<<8, 256, 0, stream>>>(preF, preB, whhw[1][0], whhw[1][1], h1);
  // 7) FC -> feats (N padded 20->64, f32 out)
  gemm_nt_wmma<<<dim3(SEQ * BATCH / 64, 1), blk, 0, stream>>>(h1, wfc, fcB, NTAGS, feats, 64, 1024, 1024, 1024);
  // 8) Viterbi decode + backtrack
  viterbi_decode<<<1, 64, 0, stream>>>(feats, trans, bp, out);
}
